// WeightedAttention_35948876267550
// MI455X (gfx1250) — compile-verified, matrix operand order checked by reading, not run on hardware
//
#include <hip/hip_runtime.h>
#include <hip/hip_bf16.h>

#define FDIM 128
#define EPSV 1e-13f

typedef __attribute__((ext_vector_type(16))) __bf16 v16bf;
typedef __attribute__((ext_vector_type(2)))  __bf16 bf2;
typedef __attribute__((ext_vector_type(8)))  float  v8f;

// ---------------- init: zero accumulators ----------------
__global__ void wa_init(float* __restrict__ y, unsigned* __restrict__ smax,
                        float* __restrict__ denom, float* __restrict__ sumc, int S) {
    size_t total = (size_t)S * FDIM;
    size_t stride = (size_t)gridDim.x * blockDim.x;
    for (size_t i = (size_t)blockIdx.x * blockDim.x + threadIdx.x; i < total; i += stride)
        y[i] = 0.0f;
    for (size_t i = (size_t)blockIdx.x * blockDim.x + threadIdx.x; i < (size_t)S; i += stride) {
        smax[i] = 0u;        // encoded -inf (below every encoded finite float)
        denom[i] = 0.0f;
        sumc[i] = 0.0f;
    }
}

// monotone float -> uint encoding for atomicMax
__device__ __forceinline__ unsigned enc_f32(float f) {
    unsigned u = __float_as_uint(f);
    return (u & 0x80000000u) ? ~u : (u | 0x80000000u);
}
__device__ __forceinline__ float dec_f32(unsigned e) {
    unsigned u = (e & 0x80000000u) ? (e & 0x7FFFFFFFu) : ~e;
    return __uint_as_float(u);
}

// ---------------- pass 1: gate logits + segment max ----------------
__global__ void wa_gate(const float* __restrict__ x, const int* __restrict__ idx,
                        const float* __restrict__ Wg, const float* __restrict__ bg,
                        float* __restrict__ g, unsigned* __restrict__ smax, int N) {
    __shared__ float sWg[FDIM];
    for (int i = threadIdx.x; i < FDIM; i += blockDim.x) sWg[i] = Wg[i];
    __syncthreads();
    int r = blockIdx.x * blockDim.x + threadIdx.x;
    if (r >= N) return;
    const float4* xr = (const float4*)(x + (size_t)r * FDIM);
    float acc = 0.0f;
#pragma unroll
    for (int j = 0; j < FDIM / 4; j++) {
        float4 v = xr[j];
        acc += v.x * sWg[4*j] + v.y * sWg[4*j+1] + v.z * sWg[4*j+2] + v.w * sWg[4*j+3];
    }
    acc += bg[0];
    g[r] = acc;
    atomicMax(&smax[idx[r]], enc_f32(acc));
}

// ---------------- pass 2: e = w*exp(g - max); denom += e ----------------
__global__ void wa_exp(const float* __restrict__ w, const int* __restrict__ idx,
                       const unsigned* __restrict__ smax, float* __restrict__ g,
                       float* __restrict__ denom, int N) {
    int r = blockIdx.x * blockDim.x + threadIdx.x;
    if (r >= N) return;
    int s = idx[r];
    float m = dec_f32(smax[s]);
    float e = w[r] * __expf(g[r] - m);
    g[r] = e;   // reuse buffer: now holds numerator e_i
    atomicAdd(&denom[s], e);
}

// ---------------- pass 3: y[s,:] += c_i * x[i,:]; sumc[s] += c_i ----------------
// Sorted index => run-length accumulate in registers, flush on segment change.
#define CHUNK 32
__global__ void wa_scatter(const float* __restrict__ x, const int* __restrict__ idx,
                           const float* __restrict__ e, const float* __restrict__ denom,
                           float* __restrict__ y, float* __restrict__ sumc, int N) {
    int gid = blockIdx.x * blockDim.x + threadIdx.x;
    int d = gid & (FDIM - 1);
    int chunk = gid >> 7;
    int r0 = chunk * CHUNK;
    if (r0 >= N) return;
    int r1 = min(r0 + CHUNK, N);
    int cur = idx[r0];
    float acc = 0.0f, accc = 0.0f;
    for (int r = r0; r < r1; r++) {
        int s = idx[r];
        if (s != cur) {
            atomicAdd(&y[(size_t)cur * FDIM + d], acc);
            if (d == 0) atomicAdd(&sumc[cur], accc);
            acc = 0.0f; accc = 0.0f; cur = s;
        }
        float c = e[r] / (denom[s] + EPSV);
        acc += c * x[(size_t)r * FDIM + d];
        if (d == 0) accc += c;
    }
    atomicAdd(&y[(size_t)cur * FDIM + d], acc);
    if (d == 0) atomicAdd(&sumc[cur], accc);
}

// ---------------- pass 4: out[s,:] = y[s,:] @ Wm + sumc[s]*bm  (in place) ----------------
// bf16 hi/lo split (3 WMMAs) for near-fp32 accuracy on v_wmma_f32_16x16x32_bf16.
__global__ void wa_gemm(float* __restrict__ y, const float* __restrict__ Wm,
                        const float* __restrict__ bm, const float* __restrict__ sumc, int S) {
    // LDS: Wm transposed, split into bf16 hi/lo; sHi[n*128+k]
    __shared__ __bf16 sHi[FDIM * FDIM];
    __shared__ __bf16 sLo[FDIM * FDIM];
    for (int i = threadIdx.x; i < FDIM * FDIM; i += blockDim.x) {
        int k = i >> 7, n = i & (FDIM - 1);
        float v = Wm[i];                 // Wm[k][n] row-major
        __bf16 h = (__bf16)v;
        __bf16 l = (__bf16)(v - (float)h);
        sHi[n * FDIM + k] = h;
        sLo[n * FDIM + k] = l;
    }
    __syncthreads();

    int lane = threadIdx.x & 31;
    int wave = threadIdx.x >> 5;
    int wavesPerBlock = blockDim.x >> 5;
    int tile = blockIdx.x * wavesPerBlock + wave;
    int numTiles = (S + 15) >> 4;
    if (tile >= numTiles) return;

    int row0 = tile * 16;
    int m    = lane & 15;
    int half = lane >> 4;

    // ---- A fragments (16x32 bf16 per k-chunk), hi/lo split ----
    v16bf ahi[4], alo[4];
    int grow = row0 + m;
    bool valid = grow < S;
    const float* yr = y + (size_t)grow * FDIM;
#pragma unroll
    for (int kc = 0; kc < 4; kc++) {
#pragma unroll
        for (int r = 0; r < 8; r++) {
            int k0 = kc * 32 + ((r < 4) ? (2 * r) : (2 * r + 8)) + half * 8;
            float2 v = valid ? *(const float2*)(yr + k0) : make_float2(0.0f, 0.0f);
            __bf16 h0 = (__bf16)v.x; __bf16 l0 = (__bf16)(v.x - (float)h0);
            __bf16 h1 = (__bf16)v.y; __bf16 l1 = (__bf16)(v.y - (float)h1);
            ahi[kc][2*r] = h0; ahi[kc][2*r+1] = h1;
            alo[kc][2*r] = l0; alo[kc][2*r+1] = l1;
        }
    }

    float scg[8];
#pragma unroll
    for (int r = 0; r < 8; r++) {
        int gm = row0 + r + half * 8;
        scg[r] = (gm < S) ? sumc[gm] : 0.0f;
    }

    int nlane = lane & 15;
#pragma unroll
    for (int nt = 0; nt < 8; nt++) {
        v8f acc = {};
        int nbase = nt * 16 + nlane;
        const __bf16* colHi = sHi + nbase * FDIM;
        const __bf16* colLo = sLo + nbase * FDIM;
#pragma unroll
        for (int kc = 0; kc < 4; kc++) {
            v16bf bhi, blo;
#pragma unroll
            for (int r = 0; r < 8; r++) {
                int k0 = kc * 32 + ((r < 4) ? (2 * r) : (2 * r + 8)) + half * 8;
                bf2 ph = *(const bf2*)(colHi + k0);
                bf2 pl = *(const bf2*)(colLo + k0);
                bhi[2*r] = ph.x; bhi[2*r+1] = ph.y;
                blo[2*r] = pl.x; blo[2*r+1] = pl.y;
            }
            acc = __builtin_amdgcn_wmma_f32_16x16x32_bf16(false, ahi[kc], false, bhi,
                                                          (short)0, acc, false, false);
            acc = __builtin_amdgcn_wmma_f32_16x16x32_bf16(false, ahi[kc], false, blo,
                                                          (short)0, acc, false, false);
            acc = __builtin_amdgcn_wmma_f32_16x16x32_bf16(false, alo[kc], false, bhi,
                                                          (short)0, acc, false, false);
        }
        float bv = bm[nbase];
#pragma unroll
        for (int r = 0; r < 8; r++) {
            int gm = row0 + r + half * 8;
            if (gm < S)
                y[(size_t)gm * FDIM + nbase] = acc[r] + scg[r] * bv;
        }
    }
}

extern "C" void kernel_launch(void* const* d_in, const int* in_sizes, int n_in,
                              void* d_out, int out_size, void* d_ws, size_t ws_size,
                              hipStream_t stream) {
    const float* x   = (const float*)d_in[0];
    const int*   idx = (const int*)  d_in[1];
    const float* w   = (const float*)d_in[2];
    const float* Wg  = (const float*)d_in[3];
    const float* bg  = (const float*)d_in[4];
    const float* Wm  = (const float*)d_in[5];
    const float* bm  = (const float*)d_in[6];
    float* out = (float*)d_out;

    int N = in_sizes[1];           // index length
    int S = out_size / FDIM;       // number of segments

    // workspace layout (floats/uints): g[N] | smax[S] | denom[S] | sumc[S]
    float*    g     = (float*)d_ws;
    unsigned* smax  = (unsigned*)(g + (size_t)N);
    float*    denom = (float*)(smax + (size_t)S);
    float*    sumc  = denom + (size_t)S;

    wa_init<<<2048, 256, 0, stream>>>(out, smax, denom, sumc, S);

    int blocksN = (N + 255) / 256;
    wa_gate<<<blocksN, 256, 0, stream>>>(x, idx, Wg, bg, g, smax, N);
    wa_exp<<<blocksN, 256, 0, stream>>>(w, idx, smax, g, denom, N);

    int chunks = (N + CHUNK - 1) / CHUNK;
    long threads = (long)chunks * FDIM;
    int blocksSc = (int)((threads + 255) / 256);
    wa_scatter<<<blocksSc, 256, 0, stream>>>(x, idx, g, denom, out, sumc, N);

    int numTiles = (S + 15) / 16;
    int wavesPerBlock = 8;                 // 256 threads
    int blocksG = (numTiles + wavesPerBlock - 1) / wavesPerBlock;
    wa_gemm<<<blocksG, 256, 0, stream>>>(out, Wm, bm, sumc, S);
}
